// RoutedMoE_20624432955923
// MI455X (gfx1250) — matrix operand order repaired and checked by reading, not compiled, not run on hardware
//
#include <hip/hip_runtime.h>

// Problem dims (reference: B=4, S=2048, D=2048, E=8, K=2, M=1024)
#define DIM   2048
#define TOK   8192         // B*S
#define NEXP  8
#define MID   1024
#define NSLOT (TOK * 2)    // T*K token copies

// GEMM tiling
#define TILE_R 64
#define TILE_N 64
#define TILE_K 32
#define LDA    40          // bf16 elems per A row in LDS (80B rows keep 16B align)

typedef __attribute__((ext_vector_type(16))) __bf16 v16bf;
typedef __attribute__((ext_vector_type(8)))  __bf16 v8bf;
typedef __attribute__((ext_vector_type(8)))  float  v8f;

__device__ __forceinline__ v16bf frag16(const __bf16* p0, const __bf16* p1) {
  v8bf lo = *(const v8bf*)p0;
  v8bf hi = *(const v8bf*)p1;
  return __builtin_shufflevector(lo, hi, 0, 1, 2, 3, 4, 5, 6, 7,
                                 8, 9, 10, 11, 12, 13, 14, 15);
}

__device__ __forceinline__ v8f wmma_bf16(v16bf a, v16bf b, v8f c) {
  // (neg_a, A, neg_b, B, c_mod, C, reuse_a, reuse_b)
  return __builtin_amdgcn_wmma_f32_16x16x32_bf16(false, a, false, b,
                                                 (short)0, c, false, false);
}

// CDNA5 async global->LDS copy (tracked by ASYNCcnt). Each lane moves 16B
// from its global address to its LDS byte offset.
__device__ __forceinline__ void async_ld_b128(unsigned lds_off,
                                              const __bf16* gptr) {
  asm volatile("global_load_async_to_lds_b128 %0, %1, off"
               :: "v"(lds_off), "v"(gptr) : "memory");
}

__device__ __forceinline__ void wait_async0() {
#if __has_builtin(__builtin_amdgcn_s_wait_asynccnt)
  __builtin_amdgcn_s_wait_asynccnt(0);
#else
  asm volatile("s_wait_asynccnt 0x0" ::: "memory");
#endif
}

// ---------------------------------------------------------------- utilities
__global__ void moe_zero_ints(int* p, int n) {
  int i = threadIdx.x;
  if (i < n) p[i] = 0;
}

__global__ __launch_bounds__(256) void moe_cvt_bf16(const float* __restrict__ in,
                                                    __bf16* __restrict__ out,
                                                    size_t n) {
  size_t i = (size_t)blockIdx.x * blockDim.x + threadIdx.x;
  size_t stride = (size_t)gridDim.x * blockDim.x;
  for (; i < n; i += stride) out[i] = (__bf16)in[i];
}

// Transposing f32->bf16 convert: in [E][R][C] -> out [E][C][R].
// grid = (C/32, R/32, E), block = 256.
__global__ __launch_bounds__(256) void moe_cvt_t(const float* __restrict__ in,
                                                 __bf16* __restrict__ out,
                                                 int R, int C) {
  __shared__ float tile[32][33];
  int e = blockIdx.z;
  int c0 = blockIdx.x * 32, r0 = blockIdx.y * 32;
  int tx = threadIdx.x & 31, ty = threadIdx.x >> 5;  // ty in 0..7
  const float* src = in + ((size_t)e * R + r0) * C + c0;
#pragma unroll
  for (int i = 0; i < 32; i += 8)
    tile[ty + i][tx] = src[(size_t)(ty + i) * C + tx];
  __syncthreads();
  __bf16* dst = out + ((size_t)e * C + c0) * R + r0;
#pragma unroll
  for (int i = 0; i < 32; i += 8)
    dst[(size_t)(ty + i) * R + tx] = (__bf16)tile[tx][ty + i];
}

// ---------------------------------------------------------------- router
__global__ __launch_bounds__(256) void moe_router(const float* __restrict__ x,
                                                  const float* __restrict__ wg,
                                                  int* __restrict__ top_idx,
                                                  float* __restrict__ top_w,
                                                  int* __restrict__ counts) {
  int wid = (blockIdx.x * 256 + threadIdx.x) >> 5;  // token id
  int lane = threadIdx.x & 31;
  if (wid >= TOK) return;
  const float* xr = x + (size_t)wid * DIM;
  float acc[NEXP];
#pragma unroll
  for (int e = 0; e < NEXP; ++e) acc[e] = 0.f;
  for (int d = lane; d < DIM; d += 32) {
    float xv = xr[d];
    const float* wr = wg + d * NEXP;
#pragma unroll
    for (int e = 0; e < NEXP; ++e) acc[e] += xv * wr[e];
  }
#pragma unroll
  for (int e = 0; e < NEXP; ++e) {
#pragma unroll
    for (int m = 16; m >= 1; m >>= 1) acc[e] += __shfl_xor(acc[e], m, 32);
  }
  if (lane == 0) {
    int i0 = 0;
    float v0 = acc[0];
#pragma unroll
    for (int e = 1; e < NEXP; ++e)
      if (acc[e] > v0) { v0 = acc[e]; i0 = e; }
    int i1 = -1;
    float v1 = -3.4e38f;
#pragma unroll
    for (int e = 0; e < NEXP; ++e)
      if (e != i0 && acc[e] > v1) { v1 = acc[e]; i1 = e; }
    float ex = __expf(v1 - v0);
    float w0 = 1.f / (1.f + ex);
    top_idx[wid * 2 + 0] = i0;
    top_idx[wid * 2 + 1] = i1;
    top_w[wid * 2 + 0] = w0;
    top_w[wid * 2 + 1] = ex * w0;
    atomicAdd(&counts[i0], 1);
    atomicAdd(&counts[i1], 1);
  }
}

__global__ void moe_prefix(const int* __restrict__ counts,
                           int* __restrict__ offsets, int* __restrict__ cursor) {
  if (threadIdx.x == 0) {
    int s = 0;
    for (int e = 0; e < NEXP; ++e) {
      offsets[e] = s;
      cursor[e] = s;
      s += counts[e];
    }
  }
}

__global__ __launch_bounds__(256) void moe_scatter(const int* __restrict__ top_idx,
                                                   const float* __restrict__ top_w,
                                                   int* __restrict__ cursor,
                                                   int* __restrict__ slot_token,
                                                   float* __restrict__ slot_w,
                                                   int* __restrict__ slot_of) {
  int t = blockIdx.x * 256 + threadIdx.x;
  if (t >= TOK) return;
#pragma unroll
  for (int k = 0; k < 2; ++k) {
    int e = top_idx[t * 2 + k];
    int s = atomicAdd(&cursor[e], 1);
    slot_token[s] = t;
    slot_w[s] = top_w[t * 2 + k];
    slot_of[t * 2 + k] = s;
  }
}

// ---------------------------------------------------------------- gate+up GEMM
// inter = silu(Xe @ wi0_e) * (Xe @ wi1_e). A (tokens) staged in LDS via async
// copies (double buffered); B fragments read from pre-transposed L2-resident
// bf16 weights, software-pipelined one K-step ahead.
__global__ __launch_bounds__(256) void moe_gate_up(
    const __bf16* __restrict__ xb, const __bf16* __restrict__ w0t,
    const __bf16* __restrict__ w1t, const int* __restrict__ slot_token,
    const int* __restrict__ offsets, const int* __restrict__ counts,
    __bf16* __restrict__ inter) {
  int e = blockIdx.z;
  int cnt = counts[e];
  int rbase = blockIdx.y * TILE_R;
  if (rbase >= cnt) return;
  int off = offsets[e];
  int nbase = blockIdx.x * TILE_N;

  __shared__ __align__(16) __bf16 lA[2][TILE_R * LDA];

  int tid = threadIdx.x;
  int wave = tid >> 5;
  int lane = tid & 31;
  int wr = wave >> 1, wc = wave & 1;  // 4 row-waves x 2 col-waves
  int hf = lane >> 4, l16 = lane & 15;

  // A loader: each thread owns one 16B chunk of the 64x32 tile
  int arow = tid >> 2;
  int akc = (tid & 3) * 8;
  int grow = rbase + arow;
  int tok = slot_token[off + (grow < cnt ? grow : cnt - 1)];
  const __bf16* aptr = xb + (size_t)tok * DIM + akc;
  unsigned ldso0 = (unsigned)(size_t)&lA[0][arow * LDA + akc];
  unsigned ldso1 = (unsigned)(size_t)&lA[1][arow * LDA + akc];

  // B fragment rows (weights are [E][MID][DIM], K contiguous per row)
  int n0 = nbase + wc * 32 + l16;
  const __bf16* b0p0 = w0t + ((size_t)e * MID + n0) * DIM + hf * 16;
  const __bf16* b0p1 = w0t + ((size_t)e * MID + n0 + 16) * DIM + hf * 16;
  const __bf16* b1p0 = w1t + ((size_t)e * MID + n0) * DIM + hf * 16;
  const __bf16* b1p1 = w1t + ((size_t)e * MID + n0 + 16) * DIM + hf * 16;

  v8f a00 = {}, a01 = {}, a10 = {}, a11 = {};

  // pipeline prologue: B frags for k0=0, async A fill of buf 0
  v16bf nb00 = frag16(b0p0, b0p0 + 8);
  v16bf nb01 = frag16(b0p1, b0p1 + 8);
  v16bf nb10 = frag16(b1p0, b1p0 + 8);
  v16bf nb11 = frag16(b1p1, b1p1 + 8);
  async_ld_b128(ldso0, aptr);

  for (int k0 = 0; k0 < DIM; k0 += TILE_K) {
    int cur = (k0 >> 5) & 1;
    v16bf cb00 = nb00, cb01 = nb01, cb10 = nb10, cb11 = nb11;
    if (k0 + TILE_K < DIM) {  // issue next-iteration B loads early
      int kn = k0 + TILE_K;
      nb00 = frag16(b0p0 + kn, b0p0 + kn + 8);
      nb01 = frag16(b0p1 + kn, b0p1 + kn + 8);
      nb10 = frag16(b1p0 + kn, b1p0 + kn + 8);
      nb11 = frag16(b1p1 + kn, b1p1 + kn + 8);
      __builtin_prefetch(b0p0 + k0 + 8 * TILE_K, 0, 3);
      __builtin_prefetch(b1p0 + k0 + 8 * TILE_K, 0, 3);
    }
    wait_async0();
    __syncthreads();
    if (k0 + TILE_K < DIM)
      async_ld_b128(cur ? ldso0 : ldso1, aptr + k0 + TILE_K);

    const __bf16* ar = &lA[cur][(wr * 16 + l16) * LDA];
    v16bf af = frag16(ar + hf * 16, ar + hf * 16 + 8);
    a00 = wmma_bf16(af, cb00, a00);
    a01 = wmma_bf16(af, cb01, a01);
    a10 = wmma_bf16(af, cb10, a10);
    a11 = wmma_bf16(af, cb11, a11);
  }

  // D layout: row = i + 8*hf within wave's 16 rows, col = l16
#pragma unroll
  for (int i = 0; i < 8; ++i) {
    int rl = rbase + wr * 16 + hf * 8 + i;
    if (rl < cnt) {
      size_t row = (size_t)(off + rl) * MID;
      float g0 = a00[i], u0 = a10[i];
      float g1 = a01[i], u1 = a11[i];
      float s0 = g0 / (1.f + __expf(-g0));
      float s1 = g1 / (1.f + __expf(-g1));
      inter[row + nbase + wc * 32 + l16] = (__bf16)(s0 * u0);
      inter[row + nbase + wc * 32 + 16 + l16] = (__bf16)(s1 * u1);
    }
  }
}

// ---------------------------------------------------------------- down GEMM
// out_sorted = router_weight * (inter_e @ wo_e); wo pre-transposed [E][DIM][MID].
__global__ __launch_bounds__(256) void moe_down(
    const __bf16* __restrict__ inter, const __bf16* __restrict__ wot,
    const float* __restrict__ slot_w, const int* __restrict__ offsets,
    const int* __restrict__ counts, float* __restrict__ outs) {
  int e = blockIdx.z;
  int cnt = counts[e];
  int rbase = blockIdx.y * TILE_R;
  if (rbase >= cnt) return;
  int off = offsets[e];
  int nbase = blockIdx.x * TILE_N;

  __shared__ __align__(16) __bf16 lA[2][TILE_R * LDA];

  int tid = threadIdx.x;
  int wave = tid >> 5;
  int lane = tid & 31;
  int wr = wave >> 1, wc = wave & 1;
  int hf = lane >> 4, l16 = lane & 15;

  int arow = tid >> 2;
  int akc = (tid & 3) * 8;
  int grow = rbase + arow;
  int asrc = off + (grow < cnt ? grow : cnt - 1);
  const __bf16* aptr = inter + (size_t)asrc * MID + akc;
  unsigned ldso0 = (unsigned)(size_t)&lA[0][arow * LDA + akc];
  unsigned ldso1 = (unsigned)(size_t)&lA[1][arow * LDA + akc];

  int n0 = nbase + wc * 32 + l16;
  const __bf16* bp0 = wot + ((size_t)e * DIM + n0) * MID + hf * 16;
  const __bf16* bp1 = wot + ((size_t)e * DIM + n0 + 16) * MID + hf * 16;

  v8f a0 = {}, a1 = {};

  v16bf nb0 = frag16(bp0, bp0 + 8);
  v16bf nb1 = frag16(bp1, bp1 + 8);
  async_ld_b128(ldso0, aptr);

  for (int k0 = 0; k0 < MID; k0 += TILE_K) {
    int cur = (k0 >> 5) & 1;
    v16bf cb0 = nb0, cb1 = nb1;
    if (k0 + TILE_K < MID) {
      int kn = k0 + TILE_K;
      nb0 = frag16(bp0 + kn, bp0 + kn + 8);
      nb1 = frag16(bp1 + kn, bp1 + kn + 8);
      __builtin_prefetch(bp0 + k0 + 8 * TILE_K, 0, 3);
      __builtin_prefetch(bp1 + k0 + 8 * TILE_K, 0, 3);
    }
    wait_async0();
    __syncthreads();
    if (k0 + TILE_K < MID)
      async_ld_b128(cur ? ldso0 : ldso1, aptr + k0 + TILE_K);

    const __bf16* ar = &lA[cur][(wr * 16 + l16) * LDA];
    v16bf af = frag16(ar + hf * 16, ar + hf * 16 + 8);
    a0 = wmma_bf16(af, cb0, a0);
    a1 = wmma_bf16(af, cb1, a1);
  }

#pragma unroll
  for (int i = 0; i < 8; ++i) {
    int rl = rbase + wr * 16 + hf * 8 + i;
    if (rl < cnt) {
      float w = slot_w[off + rl];
      size_t row = (size_t)(off + rl) * DIM;
      outs[row + nbase + wc * 32 + l16] = w * a0[i];
      outs[row + nbase + wc * 32 + 16 + l16] = w * a1[i];
    }
  }
}

// ---------------------------------------------------------------- combine
__global__ __launch_bounds__(256) void moe_combine(const float* __restrict__ outs,
                                                   const int* __restrict__ slot_of,
                                                   float* __restrict__ out) {
  int idx = blockIdx.x * 256 + threadIdx.x;  // over TOK*DIM/4 float4s
  int t = idx / (DIM / 4);
  int d4 = idx % (DIM / 4);
  int s0 = slot_of[t * 2 + 0];
  int s1 = slot_of[t * 2 + 1];
  float4 va = ((const float4*)(outs + (size_t)s0 * DIM))[d4];
  float4 vb = ((const float4*)(outs + (size_t)s1 * DIM))[d4];
  float4 r = make_float4(va.x + vb.x, va.y + vb.y, va.z + vb.z, va.w + vb.w);
  ((float4*)out)[idx] = r;
}

// ---------------------------------------------------------------- launch
extern "C" void kernel_launch(void* const* d_in, const int* in_sizes, int n_in,
                              void* d_out, int out_size, void* d_ws, size_t ws_size,
                              hipStream_t stream) {
  const float* x = (const float*)d_in[0];
  const float* wg = (const float*)d_in[1];
  const float* wi0 = (const float*)d_in[2];
  const float* wi1 = (const float*)d_in[3];
  const float* wo = (const float*)d_in[4];
  float* out = (float*)d_out;

  char* p = (char*)d_ws;
  auto carve = [&](size_t bytes) -> void* {
    void* r = (void*)p;
    p += (bytes + 255) & ~(size_t)255;
    return r;
  };
  __bf16* xb = (__bf16*)carve((size_t)TOK * DIM * 2);
  __bf16* w0t = (__bf16*)carve((size_t)NEXP * DIM * MID * 2);  // [E][MID][DIM]
  __bf16* w1t = (__bf16*)carve((size_t)NEXP * DIM * MID * 2);  // [E][MID][DIM]
  __bf16* wot = (__bf16*)carve((size_t)NEXP * MID * DIM * 2);  // [E][DIM][MID]
  __bf16* inter = (__bf16*)carve((size_t)NSLOT * MID * 2);
  float* outs = (float*)carve((size_t)NSLOT * DIM * 4);
  int* top_idx = (int*)carve((size_t)TOK * 2 * 4);
  float* top_w = (float*)carve((size_t)TOK * 2 * 4);
  int* slot_token = (int*)carve((size_t)NSLOT * 4);
  float* slot_w = (float*)carve((size_t)NSLOT * 4);
  int* slot_of = (int*)carve((size_t)NSLOT * 4);
  int* counts = (int*)carve(256);
  int* offsets = (int*)carve(256);
  int* cursor = (int*)carve(256);

  moe_zero_ints<<<1, 32, 0, stream>>>(counts, NEXP);
  moe_cvt_bf16<<<8192, 256, 0, stream>>>(x, xb, (size_t)TOK * DIM);
  // wi_0/wi_1: [E][D][M] -> [E][M][D] ; wo: [E][M][D] -> [E][D][M]
  moe_cvt_t<<<dim3(MID / 32, DIM / 32, NEXP), 256, 0, stream>>>(wi0, w0t, DIM, MID);
  moe_cvt_t<<<dim3(MID / 32, DIM / 32, NEXP), 256, 0, stream>>>(wi1, w1t, DIM, MID);
  moe_cvt_t<<<dim3(DIM / 32, MID / 32, NEXP), 256, 0, stream>>>(wo, wot, MID, DIM);
  moe_router<<<TOK / 8, 256, 0, stream>>>(x, wg, top_idx, top_w, counts);
  moe_prefix<<<1, 32, 0, stream>>>(counts, offsets, cursor);
  moe_scatter<<<TOK / 256, 256, 0, stream>>>(top_idx, top_w, cursor, slot_token,
                                             slot_w, slot_of);
  dim3 g5(MID / TILE_N, TOK / TILE_R, NEXP);
  moe_gate_up<<<g5, 256, 0, stream>>>(xb, w0t, w1t, slot_token, offsets, counts,
                                      inter);
  dim3 g6(DIM / TILE_N, TOK / TILE_R, NEXP);
  moe_down<<<g6, 256, 0, stream>>>(inter, wot, slot_w, offsets, counts, outs);
  moe_combine<<<(TOK * DIM / 4) / 256, 256, 0, stream>>>(outs, slot_of, out);
}
